// TextEncoderGenerator_6279242187447
// MI455X (gfx1250) — matrix operand order repaired and checked by reading, not compiled, not run on hardware
//
#include <hip/hip_runtime.h>
#include <hip/hip_bf16.h>

// ---------------------------------------------------------------------------
// Bidirectional GRU text encoder for MI455X (gfx1250, wave32, WMMA).
//   H=512 hidden, D=300 embed (padded 320), Z=128, B=1024, S=64.
//   * 64 batch tiles of 16 rows x 2 directions = 128 persistent blocks;
//     each block runs the whole 64-step recurrence, h in LDS (f16, double
//     buffered), per-step GEMMs via v_wmma_f32_16x16x32_f16.
//   * Weights pre-packed to the wave32 B-fragment layout; hot-loop weight
//     loads use uniform-base + 32-bit byte offsets (GVS addressing).
//   * x tile is software-pipelined: next step's rows are loaded into
//     registers during this step's GEMMs; +2-step global_prefetch_b8.
//   * Masked time-sums + fused biases live in LDS (no register blowup).
// ---------------------------------------------------------------------------

typedef _Float16 half_t;
typedef __attribute__((ext_vector_type(16))) _Float16 v16h;
typedef __attribute__((ext_vector_type(4)))  _Float16 v4h;
typedef __attribute__((ext_vector_type(8)))  float    v8f;
typedef __attribute__((ext_vector_type(4)))  float    v4f;
typedef __attribute__((ext_vector_type(8)))  unsigned v8u;

constexpr int kH   = 512;    // hidden
constexpr int kD   = 300;    // embedding
constexpr int kDP  = 320;    // embedding padded to 32
constexpr int kZ   = 128;    // latent
constexpr int kSeq = 64;     // sequence length
constexpr int kB   = 1024;   // batch

constexpr int kKTX = kDP / 32;   // 10  K-tiles for x GEMM
constexpr int kKTH = kH  / 32;   // 16  K-tiles for h GEMM
constexpr int kJT  = kH  / 16;   // 32  column tiles of H
constexpr int kNT3 = 3 * kH / 16;// 96  column tiles of 3H

constexpr int kXStride = 328;    // LDS row stride (halfs) for x tile
constexpr int kHStride = 520;    // LDS row stride (halfs) for h tile / avg tile

__device__ inline v8f splat8(float x) {
  v8f r;
#pragma unroll
  for (int i = 0; i < 8; ++i) r[i] = x;
  return r;
}

__device__ inline v8f wmma_f16(v16h a, v16h b, v8f c) {
  // D = A(16x32 f16) * B(32x16 f16) + C(16x16 f32)
  return __builtin_amdgcn_wmma_f32_16x16x32_f16(
      false, a, false, b, (short)0, c, false, false);
}

// Load a 16x32 f16 A-fragment from a row-major LDS tile (wave32
// A k order chosen here, same as pack_b_kernel: lane m=lane&15; halfpair v holds K = 16*(lane>>4) + 2*v).
__device__ inline v16h load_a_frag(const half_t* base, int strideHalfs, int kt, int lane) {
  const int m  = lane & 15;
  const int hf = (lane >> 4) & 1;
  const unsigned* u = (const unsigned*)base;
  const int rowOff = m * strideHalfs;   // strideHalfs even -> dword aligned
  v8u t;
#pragma unroll
  for (int v = 0; v < 8; ++v) {
    const int k = (kt << 5) + (hf << 4) + (v << 1);
    t[v] = u[(rowOff + k) >> 1];
  }
  return __builtin_bit_cast(v16h, t);
}

// Weight fragment load: uniform base + 32-bit byte offset (GVS addressing).
__device__ inline v16h load_b_frag(const char* base, unsigned byteOff) {
  return *(const v16h*)(base + byteOff);
}

// ---------------------------------------------------------------------------
// Pack weight W[N,K] (f32, row-major, used as B = W^T so B[k][n]=W[n][k]) into
// f16 fragment-linear layout: frag (nt,kt) is 512 halfs; lane l holds
// halfs i=0..15 with  n = 16*nt + (l&15),  k = 32*kt + 16*(l>>4) + i.
// ---------------------------------------------------------------------------
__global__ void pack_b_kernel(const float* __restrict__ src, half_t* __restrict__ dst,
                              int K, int KT, int total) {
  int e = blockIdx.x * 256 + threadIdx.x;
  if (e >= total) return;
  const int i  = e & 15;
  const int l  = (e >> 4) & 31;
  const int f  = e >> 9;           // fragment index = nt*KT + kt
  const int kt = f % KT;
  const int nt = f / KT;
  const int n  = nt * 16 + (l & 15);
  const int k  = kt * 32 + ((l >> 4) << 4) + i;
  const float v = (k < K) ? src[(size_t)n * K + k] : 0.0f;
  dst[(size_t)f * 512 + l * 16 + i] = (half_t)v;
}

// ---------------------------------------------------------------------------
// Main recurrent kernel. grid.x = 128:  blockIdx>>6 = direction, &63 = b-tile.
// 256 threads = 8 waves; wave w owns column tiles jt = 4*w .. 4*w+3 of H.
// ---------------------------------------------------------------------------
__global__ __launch_bounds__(256) void gru_seq_kernel(
    const float* __restrict__ text, const int* __restrict__ lens,
    const float* __restrict__ bih_f, const float* __restrict__ bhh_f,
    const float* __restrict__ bih_b, const float* __restrict__ bhh_b,
    const half_t* __restrict__ pWihF, const half_t* __restrict__ pWhhF,
    const half_t* __restrict__ pWihB, const half_t* __restrict__ pWhhB,
    float* __restrict__ Ssum) {

  __shared__ __attribute__((aligned(16))) half_t xs[16 * kXStride];   // 10.25 KB
  __shared__ __attribute__((aligned(16))) half_t hs[2][16 * kHStride];// 32.5 KB
  __shared__ float sums[16 * kH];                                     // 32 KB
  __shared__ float bRs[kH], bZs[kH], bINs[kH], bHNs[kH];              // 8 KB

  const int tid  = threadIdx.x;
  const int wave = tid >> 5;
  const int lane = tid & 31;
  const int hf   = (lane >> 4) & 1;
  const int nIdx = lane & 15;

  const int dir = blockIdx.x >> 6;
  const int b0  = (blockIdx.x & 63) * 16;

  const float* bih  = dir ? bih_b : bih_f;
  const float* bhh  = dir ? bhh_b : bhh_f;
  const char*  wihC = (const char*)(dir ? pWihB : pWihF);
  const char*  whhC = (const char*)(dir ? pWhhB : pWhhF);

  // Fused biases into LDS.
  for (int e = tid; e < kH; e += 256) {
    bRs[e]  = bih[e]          + bhh[e];
    bZs[e]  = bih[kH + e]     + bhh[kH + e];
    bINs[e] = bih[2 * kH + e];
    bHNs[e] = bhh[2 * kH + e];
  }
  // sums = 0, h0 = 0, xs = 0 (pad columns >= 300 stay zero forever).
  for (int e = tid; e < 16 * kH; e += 256) sums[e] = 0.0f;
  for (int e = tid; e < 16 * kHStride; e += 256) hs[0][e] = (half_t)0.0f;
  for (int e = tid; e < 16 * kXStride; e += 256) xs[e] = (half_t)0.0f;

  // Row lengths for this wave's fragment rows (m = i + 8*hf).
  int lenr[8];
#pragma unroll
  for (int i = 0; i < 8; ++i) lenr[i] = lens[b0 + i + 8 * hf];

  // x staging map: thread -> (row = tid>>4, 20 contiguous cols at (tid&15)*20).
  // Threads with colBase == 300 only ever hold the zero padding.
  const int  xrow   = tid >> 4;
  const int  cb     = (tid & 15) * 20;
  const bool xvalid = (cb < kD);
  const int  xsBase = xrow * kXStride + cb;                    // halfs, 8B aligned
  const float* tRow = text + ((size_t)(b0 + xrow) * kSeq) * kD + cb;

  auto xIdx = [&](int s) -> int { return dir ? (kSeq - 1 - s) : s; };

  // Pipeline buffer: next step's 20 x values live in registers.
  v4f xv[5];
  if (xvalid) {
    const float* p = tRow + (size_t)xIdx(0) * kD;
#pragma unroll
    for (int q = 0; q < 5; ++q) xv[q] = *(const v4f*)(p + 4 * q);
  }

  for (int s = 0; s < kSeq; ++s) {
    __syncthreads();  // prev step consumed xs; hs[s&1] complete

    // Commit the pipelined x tile (f32 regs -> f16 LDS), 5x ds_store_b64.
    if (xvalid) {
#pragma unroll
      for (int q = 0; q < 5; ++q) {
        v4h h4;
#pragma unroll
        for (int c = 0; c < 4; ++c) h4[c] = (half_t)xv[q][c];
        *(v4h*)(xs + xsBase + 4 * q) = h4;
      }
    }
    __syncthreads();

    // Issue next step's x loads now; results consumed next iteration.
    if (xvalid && s + 1 < kSeq) {
      const float* p = tRow + (size_t)xIdx(s + 1) * kD;
#pragma unroll
      for (int q = 0; q < 5; ++q) xv[q] = *(const v4f*)(p + 4 * q);
    }
    // Pull step s+2 toward L2 (global_prefetch_b8).
    if (xvalid && s + 2 < kSeq)
      __builtin_prefetch(tRow + (size_t)xIdx(s + 2) * kD, 0, 1);

    const half_t* hcur  = hs[s & 1];
    half_t*       hnext = hs[(s + 1) & 1];

#pragma unroll 1
    for (int j = 0; j < 4; ++j) {
      const int jt  = wave * 4 + j;
      const int col = jt * 16 + nIdx;

      v8f accR  = splat8(bRs[col]);
      v8f accZ  = splat8(bZs[col]);
      v8f accIN = splat8(bINs[col]);
      v8f accHN = splat8(bHNs[col]);

      // 32-bit byte offsets: fragment (nt,kt) sits at (nt*KT + kt)*1024 + lane*32.
      const unsigned laneOff = (unsigned)lane * 32u;
      const unsigned xo0 = (unsigned)(((0 * kJT + jt) * kKTX) << 10) + laneOff;
      const unsigned xo1 = (unsigned)(((1 * kJT + jt) * kKTX) << 10) + laneOff;
      const unsigned xo2 = (unsigned)(((2 * kJT + jt) * kKTX) << 10) + laneOff;

      // gi = x @ Wih^T
#pragma unroll 2
      for (int kt = 0; kt < kKTX; ++kt) {
        const unsigned ko = (unsigned)kt << 10;
        const v16h a  = load_a_frag(xs, kXStride, kt, lane);
        const v16h w0 = load_b_frag(wihC, xo0 + ko);
        const v16h w1 = load_b_frag(wihC, xo1 + ko);
        const v16h w2 = load_b_frag(wihC, xo2 + ko);
        accR  = wmma_f16(a, w0, accR);
        accZ  = wmma_f16(a, w1, accZ);
        accIN = wmma_f16(a, w2, accIN);
      }

      const unsigned ho0 = (unsigned)(((0 * kJT + jt) * kKTH) << 10) + laneOff;
      const unsigned ho1 = (unsigned)(((1 * kJT + jt) * kKTH) << 10) + laneOff;
      const unsigned ho2 = (unsigned)(((2 * kJT + jt) * kKTH) << 10) + laneOff;

      // gh = h @ Whh^T
#pragma unroll 2
      for (int kt = 0; kt < kKTH; ++kt) {
        const unsigned ko = (unsigned)kt << 10;
        const v16h a  = load_a_frag(hcur, kHStride, kt, lane);
        const v16h w0 = load_b_frag(whhC, ho0 + ko);
        const v16h w1 = load_b_frag(whhC, ho1 + ko);
        const v16h w2 = load_b_frag(whhC, ho2 + ko);
        accR  = wmma_f16(a, w0, accR);
        accZ  = wmma_f16(a, w1, accZ);
        accHN = wmma_f16(a, w2, accHN);
      }

      // Gate math on the C-fragment layout (VGPR i -> row m=i+8*hf, col).
#pragma unroll
      for (int i = 0; i < 8; ++i) {
        const int m    = i + 8 * hf;
        const float hold = (float)hcur[m * kHStride + col];
        const float r = 1.0f / (1.0f + __expf(-accR[i]));
        const float z = 1.0f / (1.0f + __expf(-accZ[i]));
        const float n = tanhf(accIN[i] + r * accHN[i]);
        const float hn = (1.0f - z) * n + z * hold;      // carry is UNmasked
        const float add = (s < lenr[i]) ? hn : 0.0f;     // output sum is masked
        sums[m * kH + col] += add;                       // wave-exclusive slot
        hnext[m * kHStride + col] = (half_t)hn;
      }
    }
  }

  __syncthreads();
  // Write running sums: Ssum[dir][b][h]
  float* out = Ssum + (size_t)dir * kB * kH;
  for (int e = tid; e < 16 * kH; e += 256) {
    const int row = e >> 9, col = e & (kH - 1);
    out[(size_t)(b0 + row) * kH + col] = sums[e];
  }
}

// ---------------------------------------------------------------------------
// Final kernel: avg -> (mu, sigma) heads via WMMA -> reparam sample.
// grid.x = 64 (batch tiles of 16), 128 threads (4 waves, 2 N-tiles each).
// d_out = [final(1024x128) | mu | sigma], f32.
// ---------------------------------------------------------------------------
__global__ __launch_bounds__(128) void heads_kernel(
    const float* __restrict__ Ssum, const int* __restrict__ lens,
    const half_t* __restrict__ pWmu, const half_t* __restrict__ pWsig,
    const float* __restrict__ b_mu, const float* __restrict__ b_sig,
    const float* __restrict__ eps, float* __restrict__ out) {

  __shared__ __attribute__((aligned(16))) half_t avgs[16 * kHStride];

  const int tid  = threadIdx.x;
  const int wave = tid >> 5;
  const int lane = tid & 31;
  const int hf   = (lane >> 4) & 1;
  const int nIdx = lane & 15;
  const int b0   = blockIdx.x * 16;

  const float* Sf = Ssum;
  const float* Sb = Ssum + (size_t)kB * kH;

  for (int e = tid; e < 16 * kH; e += 128) {
    const int row = e >> 9, col = e & (kH - 1);
    const float inv = 1.0f / (float)lens[b0 + row];
    const float v = 0.5f * (Sf[(size_t)(b0 + row) * kH + col] +
                            Sb[(size_t)(b0 + row) * kH + col]) * inv;
    avgs[row * kHStride + col] = (half_t)v;
  }
  __syncthreads();

  const char* wmuC  = (const char*)pWmu;
  const char* wsigC = (const char*)pWsig;
  float* oFin = out;
  float* oMu  = out + (size_t)kB * kZ;
  float* oSig = out + 2 * (size_t)kB * kZ;

#pragma unroll 1
  for (int t = 0; t < 2; ++t) {
    const int nt = wave * 2 + t;             // 8 N-tiles cover Z=128
    v8f am = splat8(b_mu[nt * 16 + nIdx]);
    v8f as = splat8(b_sig[nt * 16 + nIdx]);
    const unsigned base = (unsigned)((nt * kKTH) << 10) + (unsigned)lane * 32u;
#pragma unroll 2
    for (int kt = 0; kt < kKTH; ++kt) {
      const unsigned ko = base + ((unsigned)kt << 10);
      const v16h a  = load_a_frag(avgs, kHStride, kt, lane);
      const v16h wm = load_b_frag(wmuC,  ko);
      const v16h ws = load_b_frag(wsigC, ko);
      am = wmma_f16(a, wm, am);
      as = wmma_f16(a, ws, as);
    }
#pragma unroll
    for (int i = 0; i < 8; ++i) {
      const int row = b0 + i + 8 * hf;
      const int col = nt * 16 + nIdx;
      float mu = am[i];  mu = (mu >= 0.0f) ? mu : 0.2f * mu;        // leaky relu
      float sg = as[i];  sg = (sg >= 0.0f) ? sg : 0.2f * sg;
      const float sigma = __expf(sg);
      const float ep = eps[(size_t)row * kZ + col];
      oFin[(size_t)row * kZ + col] = ep * sigma + mu;
      oMu [(size_t)row * kZ + col] = mu;
      oSig[(size_t)row * kZ + col] = sigma;
    }
  }
}

// ---------------------------------------------------------------------------
extern "C" void kernel_launch(void* const* d_in, const int* in_sizes, int n_in,
                              void* d_out, int out_size, void* d_ws, size_t ws_size,
                              hipStream_t stream) {
  (void)in_sizes; (void)n_in; (void)out_size; (void)ws_size;

  const float* text  = (const float*)d_in[0];
  const int*   lens  = (const int*)d_in[1];
  const float* Wih_f = (const float*)d_in[2];
  const float* Whh_f = (const float*)d_in[3];
  const float* bih_f = (const float*)d_in[4];
  const float* bhh_f = (const float*)d_in[5];
  const float* Wih_b = (const float*)d_in[6];
  const float* Whh_b = (const float*)d_in[7];
  const float* bih_b = (const float*)d_in[8];
  const float* bhh_b = (const float*)d_in[9];
  const float* W_mu  = (const float*)d_in[10];
  const float* b_mu  = (const float*)d_in[11];
  const float* W_sig = (const float*)d_in[12];
  const float* b_sig = (const float*)d_in[13];
  const float* eps   = (const float*)d_in[14];

  // Workspace carve-up (~9.6 MB total).
  char* ws = (char*)d_ws;
  size_t off = 0;
  auto take = [&](size_t bytes) -> char* {
    char* p = ws + off;
    off = (off + bytes + 255) & ~(size_t)255;
    return p;
  };
  const int nIH = kNT3 * kKTX * 512;      // 491520 halfs per Wih
  const int nHH = kNT3 * kKTH * 512;      // 786432 halfs per Whh
  const int nHD = (kZ / 16) * kKTH * 512; // 65536 halfs per head weight
  half_t* pWihF = (half_t*)take((size_t)nIH * 2);
  half_t* pWhhF = (half_t*)take((size_t)nHH * 2);
  half_t* pWihB = (half_t*)take((size_t)nIH * 2);
  half_t* pWhhB = (half_t*)take((size_t)nHH * 2);
  half_t* pWmu  = (half_t*)take((size_t)nHD * 2);
  half_t* pWsig = (half_t*)take((size_t)nHD * 2);
  float*  Ssum  = (float*)take((size_t)2 * kB * kH * 4);

  // Weight packing (f32 -> f16 WMMA B-fragment layout).
  pack_b_kernel<<<(nIH + 255) / 256, 256, 0, stream>>>(Wih_f, pWihF, kD, kKTX, nIH);
  pack_b_kernel<<<(nHH + 255) / 256, 256, 0, stream>>>(Whh_f, pWhhF, kH, kKTH, nHH);
  pack_b_kernel<<<(nIH + 255) / 256, 256, 0, stream>>>(Wih_b, pWihB, kD, kKTX, nIH);
  pack_b_kernel<<<(nHH + 255) / 256, 256, 0, stream>>>(Whh_b, pWhhB, kH, kKTH, nHH);
  pack_b_kernel<<<(nHD + 255) / 256, 256, 0, stream>>>(W_mu,  pWmu,  kH, kKTH, nHD);
  pack_b_kernel<<<(nHD + 255) / 256, 256, 0, stream>>>(W_sig, pWsig, kH, kKTH, nHD);

  // Recurrence: 64 batch tiles x 2 directions, 256 threads (8 waves) each.
  gru_seq_kernel<<<dim3(128), dim3(256), 0, stream>>>(
      text, lens, bih_f, bhh_f, bih_b, bhh_b,
      pWihF, pWhhF, pWihB, pWhhB, Ssum);

  // Heads + reparameterized sample.
  heads_kernel<<<dim3(64), dim3(128), 0, stream>>>(
      Ssum, lens, pWmu, pWsig, b_mu, b_sig, eps, (float*)d_out);
}
